// Attention_52776558133444
// MI455X (gfx1250) — matrix operand order repaired and checked by reading, not compiled, not run on hardware
//
#include <hip/hip_runtime.h>

// ---------------------------------------------------------------- constants
#define B_    2
#define S_    2048
#define D_    2048
#define QH_   32
#define KVH_  8
#define HD_   64
#define GRP_  4   // QH_/KVH_

typedef __bf16 bf16_t;
typedef __attribute__((ext_vector_type(16))) __bf16 v16bf;
typedef __attribute__((ext_vector_type(8)))  float  v8f;
typedef __attribute__((ext_vector_type(4)))  float  f32x4;
typedef int v4i __attribute__((vector_size(16)));   // matches builtin param type

#define AS1 __attribute__((address_space(1)))
#define AS3 __attribute__((address_space(3)))

// Async global->LDS staging (CDNA5: global_load_async_to_lds_b128, ASYNCcnt)
#if defined(__has_builtin)
#if __has_builtin(__builtin_amdgcn_global_load_async_to_lds_b128) && \
    __has_builtin(__builtin_amdgcn_s_wait_asynccnt)
#define USE_ASYNC_LDS 1
#endif
#endif
#ifndef USE_ASYNC_LDS
#define USE_ASYNC_LDS 0
#endif

__device__ __forceinline__ AS1 v4i* glob_v4(const void* p) {
  return (AS1 v4i*)(unsigned long long)(uintptr_t)p;   // flat==global vaddr
}
__device__ __forceinline__ AS3 v4i* lds_v4(void* p) {
  return (AS3 v4i*)(unsigned int)(uintptr_t)p;         // low 32b = LDS offset
}

// stage 32 bytes (16 bf16) global -> LDS
__device__ __forceinline__ void stage32(const bf16_t* g, bf16_t* l) {
#if USE_ASYNC_LDS
  __builtin_amdgcn_global_load_async_to_lds_b128(glob_v4(g),     lds_v4(l),     0, 0);
  __builtin_amdgcn_global_load_async_to_lds_b128(glob_v4(g + 8), lds_v4(l + 8), 0, 0);
#else
  *(f32x4*)l       = *(const f32x4*)g;
  *(f32x4*)(l + 8) = *(const f32x4*)(g + 8);
#endif
}
// wait own asyncs (or nothing), then block barrier -> tiles visible to all
__device__ __forceinline__ void stage_commit() {
#if USE_ASYNC_LDS
  __builtin_amdgcn_s_wait_asynccnt(0);
#endif
  __syncthreads();
}

// ---------------------------------------------------------------- WMMA utils
__device__ __forceinline__ v8f wmma_bf16(v16bf a, v16bf b, v8f c) {
  return __builtin_amdgcn_wmma_f32_16x16x32_bf16(
      false, a, false, b, (short)0, c, false, false);
}

// A-fragment (16x32 MxK row-major, row stride lda elems).
// lanes 0-15: M=lane, K=0..15; lanes 16-31: K=16..31.
__device__ __forceinline__ v16bf load_frag_a(const bf16_t* base, int lda) {
  const int lane = threadIdx.x & 31;
  const bf16_t* p = base + (lane & 15) * lda + (lane >> 4) * 16;
  union { v16bf v; f32x4 f[2]; } u;
  u.f[0] = *(const f32x4*)(p);
  u.f[1] = *(const f32x4*)(p + 8);
  return u.v;
}

// B-fragment from B^T (N x K row-major, row stride ldb elems).
// lane holds column N=lane&15, contiguous K chunk at 16*(lane>>4).
__device__ __forceinline__ v16bf load_frag_bt(const bf16_t* base, int ldb) {
  const int lane = threadIdx.x & 31;
  const bf16_t* p = base + (lane & 15) * ldb + (lane >> 4) * 16;
  union { v16bf v; f32x4 f[2]; } u;
  u.f[0] = *(const f32x4*)(p);
  u.f[1] = *(const f32x4*)(p + 8);
  return u.v;
}

// ---------------------------------------------------------------- converts
__global__ void cvt_kernel(const float* __restrict__ src, bf16_t* __restrict__ dst, int n) {
  int i = blockIdx.x * blockDim.x + threadIdx.x;
  if (i < n) dst[i] = (bf16_t)src[i];
}

// dst[n*K + k] = src[k*N + n]  (W [K,N] -> W^T bf16 [N,K])
__global__ void transpose_cvt_kernel(const float* __restrict__ src, bf16_t* __restrict__ dst,
                                     int K, int N) {
  int i = blockIdx.x * blockDim.x + threadIdx.x;
  if (i >= K * N) return;
  int n = i % N;
  int k = i / N;
  dst[(size_t)n * K + k] = (bf16_t)src[(size_t)k * N + n];
}

// ---------------------------------------------------------------- GEMM
// C[M,N] f32 = A[M,K] bf16 @ (Bt[N,K])^T + bias[N]
// 256 threads (8 waves); block tile 128x128; wave tile 32x64; K step 32.
__global__ void __launch_bounds__(256)
gemm_bf16_kernel(const bf16_t* __restrict__ A, const bf16_t* __restrict__ Bt,
                 const float* __restrict__ bias, float* __restrict__ C,
                 int M, int N, int K) {
  __shared__ bf16_t sA[128 * 32];   // 8 KB
  __shared__ bf16_t sB[128 * 32];   // 8 KB
  const int m0   = blockIdx.y * 128;
  const int n0   = blockIdx.x * 128;
  const int t    = threadIdx.x;
  const int wave = t >> 5;
  const int lane = t & 31;
  const int wm   = wave >> 1;        // 0..3  (32-row strip)
  const int wn   = wave & 1;         // 0..1  (64-col strip)
  const int lrow = t >> 1;           // 0..127 staging row
  const int lcol = (t & 1) * 16;     // 0/16 staging chunk

  v8f acc[2][4] = {{{}, {}, {}, {}}, {{}, {}, {}, {}}};

  for (int k0 = 0; k0 < K; k0 += 32) {
    const bf16_t* ga = A  + (size_t)(m0 + lrow) * K + k0 + lcol;
    const bf16_t* gb = Bt + (size_t)(n0 + lrow) * K + k0 + lcol;
    if (k0 + 32 < K) {
      __builtin_prefetch(ga + 32, 0, 1);
      __builtin_prefetch(gb + 32, 0, 1);
    }
    __syncthreads();                       // prior-iteration fragment reads done
    stage32(ga, sA + lrow * 32 + lcol);    // async b128 pair (or sync fallback)
    stage32(gb, sB + lrow * 32 + lcol);
    stage_commit();                        // s_wait_asynccnt 0 + barrier

    v16bf af0 = load_frag_a(sA + (wm * 32 + 0)  * 32, 32);
    v16bf af1 = load_frag_a(sA + (wm * 32 + 16) * 32, 32);
#pragma unroll
    for (int n = 0; n < 4; ++n) {
      v16bf bf = load_frag_bt(sB + (wn * 64 + n * 16) * 32, 32);
      acc[0][n] = wmma_bf16(af0, bf, acc[0][n]);
      acc[1][n] = wmma_bf16(af1, bf, acc[1][n]);
    }
  }

  const int kh = lane >> 4, nn = lane & 15;
#pragma unroll
  for (int n = 0; n < 4; ++n) {
    const int col = n0 + wn * 64 + n * 16 + nn;
    const float bv = bias ? bias[col] : 0.0f;
#pragma unroll
    for (int i = 0; i < 2; ++i)
#pragma unroll
      for (int r = 0; r < 8; ++r) {
        const int row = m0 + wm * 32 + i * 16 + r + 8 * kh;
        C[(size_t)row * N + col] = acc[i][n][r] + bv;
      }
  }
}

// ---------------------------------------------------------------- RoPE pack
// src: [B*S, H*64] f32 ; freqs: [S,32,2] f32 ; dst: [B,H,S,64] bf16 (rotated)
__global__ void rope_pack_kernel(const float* __restrict__ src,
                                 const float* __restrict__ freqs,
                                 bf16_t* __restrict__ dst, int H) {
  int i = blockIdx.x * blockDim.x + threadIdx.x;
  int total = B_ * S_ * H * 32;
  if (i >= total) return;
  int p = i & 31;
  int h = (i >> 5) % H;
  int s = ((i >> 5) / H) % S_;
  int b = i / (32 * H * S_);
  const float* sp = src + ((size_t)(b * S_ + s) * H + h) * 64 + p * 2;
  float x0 = sp[0], x1 = sp[1];
  float c  = freqs[(s * 32 + p) * 2 + 0];
  float sn = freqs[(s * 32 + p) * 2 + 1];
  bf16_t* dp = dst + (((size_t)b * H + h) * S_ + s) * 64 + p * 2;
  dp[0] = (bf16_t)(x0 * c - x1 * sn);
  dp[1] = (bf16_t)(x0 * sn + x1 * c);
}

// src: [B*S, KVH*64] f32 -> dst: [B,KVH,64,S] bf16 (transposed for PV B-operand)
__global__ void pack_v_kernel(const float* __restrict__ src, bf16_t* __restrict__ dst) {
  int i = blockIdx.x * blockDim.x + threadIdx.x;
  int total = B_ * S_ * KVH_ * 64;
  if (i >= total) return;
  int d = i & 63;
  int h = (i >> 6) % KVH_;
  int s = ((i >> 6) / KVH_) % S_;
  int b = i / (64 * KVH_ * S_);
  dst[(((size_t)b * KVH_ + h) * 64 + d) * S_ + s] =
      (bf16_t)src[((size_t)(b * S_ + s) * KVH_ + h) * 64 + d];
}

// ---------------------------------------------------------------- attention
// q: [B,QH,S,64] (roped), k: [B,KVH,S,64] (roped), v: [B,KVH,64,S], all bf16
// o: [B,S,QH*64] bf16 (token-major for the output projection)
// Block = 4 waves = 64 queries of ONE head; K/V tiles staged in LDS per block.
__global__ void __launch_bounds__(128)
attn_kernel(const bf16_t* __restrict__ q, const bf16_t* __restrict__ k,
            const bf16_t* __restrict__ v, bf16_t* __restrict__ o) {
  __shared__ bf16_t sK[32 * 64];      // [key][hd]   4 KB
  __shared__ bf16_t sV[64 * 32];      // [hd][key]   4 KB
  __shared__ bf16_t sP[4][16 * 32];   // per-wave P  4 KB
  const int t    = threadIdx.x;
  const int wave = t >> 5;
  const int lane = t & 31;

  const int head = blockIdx.x >> 5;               // 32 blocks per head
  const int qt   = (blockIdx.x & 31) * 4 + wave;  // 16-query tile, 0..127
  const int qh   = head % QH_;
  const int b    = head / QH_;
  const int kvh  = qh / GRP_;

  const bf16_t* qp = q + ((size_t)head * S_ + qt * 16) * 64;
  const bf16_t* kp = k + (size_t)(b * KVH_ + kvh) * S_ * 64;   // [S,64]
  const bf16_t* vp = v + (size_t)(b * KVH_ + kvh) * 64 * S_;   // [64,S]

  const v16bf a0 = load_frag_a(qp, 64);        // hd 0..31
  const v16bf a1 = load_frag_a(qp + 32, 64);   // hd 32..63

  v8f oacc[4] = {{}, {}, {}, {}};
  float m[8], l[8];
#pragma unroll
  for (int r = 0; r < 8; ++r) { m[r] = -INFINITY; l[r] = 0.0f; }

  const int kh = lane >> 4, nn = lane & 15;
  bf16_t* P = sP[wave];

  // staging split: K tile row=t>>2 (32 rows x 64), V tile row=t>>1 (64 rows x 32)
  const int krow = t >> 2, kcol = (t & 3) * 16;
  const int vrow = t >> 1, vcol = (t & 1) * 16;

  for (int key0 = 0; key0 < S_; key0 += 32) {
    __syncthreads();                                   // prior tile reads done
    stage32(kp + (size_t)(key0 + krow) * 64 + kcol, sK + krow * 64 + kcol);
    stage32(vp + (size_t)vrow * S_ + key0 + vcol,   sV + vrow * 32 + vcol);
    stage_commit();

    // ---- scores: Q(16x64) @ K^T -> two 16x16 tiles
    v8f sc[2];
#pragma unroll
    for (int j = 0; j < 2; ++j) {
      v8f c = {};
      c = wmma_bf16(a0, load_frag_bt(sK + j * 16 * 64, 64), c);
      c = wmma_bf16(a1, load_frag_bt(sK + j * 16 * 64 + 32, 64), c);
      sc[j] = c;
    }
    // ---- online softmax (rows live in 16-lane halves)
#pragma unroll
    for (int r = 0; r < 8; ++r) {
      float s0 = sc[0][r] * 0.125f;                    // 1/sqrt(64)
      float s1 = sc[1][r] * 0.125f;
      float rm = fmaxf(s0, s1);
#pragma unroll
      for (int w = 1; w < 16; w <<= 1) rm = fmaxf(rm, __shfl_xor(rm, w, 32));
      float nm    = fmaxf(m[r], rm);
      float scale = __expf(m[r] - nm);
      m[r] = nm;
      float p0 = __expf(s0 - nm);
      float p1 = __expf(s1 - nm);
      float ps = p0 + p1;
#pragma unroll
      for (int w = 1; w < 16; w <<= 1) ps += __shfl_xor(ps, w, 32);
      l[r] = l[r] * scale + ps;
#pragma unroll
      for (int n = 0; n < 4; ++n) oacc[n][r] *= scale;
      const int row = r + 8 * kh;
      P[row * 32 + nn]      = (bf16_t)p0;              // wave-private region:
      P[row * 32 + 16 + nn] = (bf16_t)p1;              // dscnt wait suffices
    }
    // ---- O += P(16x32) @ V(32x64)
    v16bf pf = load_frag_a(P, 32);
#pragma unroll
    for (int n = 0; n < 4; ++n)
      oacc[n] = wmma_bf16(pf, load_frag_bt(sV + n * 16 * 32, 32), oacc[n]);
  }

  // ---- epilogue: normalize, store token-major bf16
#pragma unroll
  for (int r = 0; r < 8; ++r) {
    const float inv   = 1.0f / l[r];
    const int   token = qt * 16 + r + 8 * kh;
    const size_t rowoff = ((size_t)b * S_ + token) * (QH_ * HD_);
#pragma unroll
    for (int n = 0; n < 4; ++n)
      o[rowoff + qh * 64 + n * 16 + nn] = (bf16_t)(oacc[n][r] * inv);
  }
}

// ---------------------------------------------------------------- launcher
extern "C" void kernel_launch(void* const* d_in, const int* in_sizes, int n_in,
                              void* d_out, int out_size, void* d_ws, size_t ws_size,
                              hipStream_t stream) {
  const float* x     = (const float*)d_in[0];
  const float* freqs = (const float*)d_in[1];
  const float* Wq = (const float*)d_in[2];
  const float* bq = (const float*)d_in[3];
  const float* Wk = (const float*)d_in[4];
  const float* bk = (const float*)d_in[5];
  const float* Wv = (const float*)d_in[6];
  const float* bv = (const float*)d_in[7];
  const float* Wo = (const float*)d_in[8];
  const float* bo = (const float*)d_in[9];
  float* out = (float*)d_out;

  char* ws = (char*)d_ws;
  constexpr size_t MB = (size_t)1 << 20;
  bf16_t* xb  = (bf16_t*)(ws + 0 * MB);    // 16 MB
  bf16_t* WqT = (bf16_t*)(ws + 16 * MB);   //  8 MB
  bf16_t* WkT = (bf16_t*)(ws + 24 * MB);   //  2 MB
  bf16_t* WvT = (bf16_t*)(ws + 26 * MB);   //  2 MB
  bf16_t* WoT = (bf16_t*)(ws + 28 * MB);   //  8 MB
  float*  qf  = (float*) (ws + 36 * MB);   // 32 MB (dead after rope)
  float*  kf  = (float*) (ws + 68 * MB);   //  8 MB
  float*  vf  = (float*) (ws + 76 * MB);   //  8 MB
  bf16_t* qb  = (bf16_t*)(ws + 84 * MB);   // 16 MB
  bf16_t* kb  = (bf16_t*)(ws + 100 * MB);  //  4 MB
  bf16_t* vb  = (bf16_t*)(ws + 104 * MB);  //  4 MB  -> total 108 MB
  bf16_t* ob  = (bf16_t*)(ws + 36 * MB);   // 16 MB aliases dead qf

  cvt_kernel<<<(B_ * S_ * D_) / 256, 256, 0, stream>>>(x, xb, B_ * S_ * D_);
  transpose_cvt_kernel<<<(2048 * 2048) / 256, 256, 0, stream>>>(Wq, WqT, 2048, 2048);
  transpose_cvt_kernel<<<(2048 * 512) / 256, 256, 0, stream>>>(Wk, WkT, 2048, 512);
  transpose_cvt_kernel<<<(2048 * 512) / 256, 256, 0, stream>>>(Wv, WvT, 2048, 512);
  transpose_cvt_kernel<<<(2048 * 2048) / 256, 256, 0, stream>>>(Wo, WoT, 2048, 2048);

  gemm_bf16_kernel<<<dim3(2048 / 128, 4096 / 128), 256, 0, stream>>>(xb, WqT, bq, qf, 4096, 2048, 2048);
  gemm_bf16_kernel<<<dim3(512 / 128, 4096 / 128), 256, 0, stream>>>(xb, WkT, bk, kf, 4096, 512, 2048);
  gemm_bf16_kernel<<<dim3(512 / 128, 4096 / 128), 256, 0, stream>>>(xb, WvT, bv, vf, 4096, 512, 2048);

  rope_pack_kernel<<<(B_ * S_ * QH_ * 32) / 256, 256, 0, stream>>>(qf, freqs, qb, QH_);
  rope_pack_kernel<<<(B_ * S_ * KVH_ * 32) / 256, 256, 0, stream>>>(kf, freqs, kb, KVH_);
  pack_v_kernel<<<(B_ * S_ * KVH_ * 64) / 256, 256, 0, stream>>>(vf, vb);

  attn_kernel<<<B_ * QH_ * 32, 128, 0, stream>>>(qb, kb, vb, ob);

  gemm_bf16_kernel<<<dim3(2048 / 128, 4096 / 128), 256, 0, stream>>>(ob, WoT, bo, out, 4096, 2048, 2048);
}
